// SlotAttentionBase_26053271617985
// MI455X (gfx1250) — matrix operand order repaired and checked by reading, not compiled, not run on hardware
//
#include <hip/hip_runtime.h>

#define BB 64
#define NN 4096
#define DD 256
#define NSL 8
#define HIDD 512
#define SCALE 0.0625f
#define LN_EPS 1e-5f
#define EPS 1e-8f

typedef __bf16 bf16;
typedef __attribute__((ext_vector_type(16))) __bf16 v16bf;
typedef __attribute__((ext_vector_type(8)))  __bf16 bf16x8;
typedef __attribute__((ext_vector_type(8)))  float  v8f;

// ---------------------------------------------------------------------------
// D = A(16x32 bf16) x B(32x16 bf16) + C(16x16 f32)   (wave32 WMMA)
// ---------------------------------------------------------------------------
static __device__ inline v8f wmma_bf16(v16bf a, v16bf b, v8f c) {
    return __builtin_amdgcn_wmma_f32_16x16x32_bf16(
        /*neg_a=*/false, a, /*neg_b=*/false, b,
        /*c_mod=*/(short)0, c, /*reuse_a=*/false, /*reuse_b=*/false);
}

// Assemble a v16bf operand from two contiguous 8-element (16-byte) chunks.
static __device__ inline v16bf ldpair(const bf16* lo, const bf16* hi) {
    bf16x8 l = *(const bf16x8*)lo;
    bf16x8 h = *(const bf16x8*)hi;
    v16bf r;
#pragma unroll
    for (int i = 0; i < 8; i++) { r[i] = l[i]; r[i + 8] = h[i]; }
    return r;
}

// 256-thread block sum (8 waves of 32). `red` must hold >= 8 floats.
static __device__ inline float blockSum(float v, float* red) {
    int lane = threadIdx.x & 31;
    int w = threadIdx.x >> 5;
#pragma unroll
    for (int o = 16; o > 0; o >>= 1) v += __shfl_xor(v, o, 32);
    if (lane == 0) red[w] = v;
    __syncthreads();
    if (threadIdx.x == 0) {
        float t = 0.f;
#pragma unroll
        for (int i = 0; i < 8; i++) t += red[i];
        red[0] = t;
    }
    __syncthreads();
    float r = red[0];
    __syncthreads();
    return r;
}

// ---------------------------------------------------------------------------
// 0) Transpose+convert Wk / Wv to bf16 column-major (WT[n][k] = W[k][n])
// ---------------------------------------------------------------------------
__global__ __launch_bounds__(256) void prep_transpose_kernel(
    const float* __restrict__ Wk, const float* __restrict__ Wv,
    bf16* __restrict__ WkT, bf16* __restrict__ WvT) {
    int n = blockIdx.x & 255;
    int selv = blockIdx.x >> 8;
    const float* W = selv ? Wv : Wk;
    bf16* WT = selv ? WvT : WkT;
    int k = threadIdx.x;
    WT[n * DD + k] = (bf16)W[k * DD + n];
}

// ---------------------------------------------------------------------------
// 1) slots = mu + exp(logsigma) * noise
// ---------------------------------------------------------------------------
__global__ __launch_bounds__(256) void init_slots_kernel(
    const float* __restrict__ noise, const float* __restrict__ mu,
    const float* __restrict__ lsig, float* __restrict__ slots) {
    int idx = blockIdx.x * 256 + threadIdx.x;
    int d = idx & (DD - 1);
    slots[idx] = mu[d] + __expf(lsig[d]) * noise[idx];
}

// ---------------------------------------------------------------------------
// 2) Fused layernorm(inputs) + dual GEMM via WMMA:
//      k  = x@Wk  stored row-major      k[b][token][feat]   (B operand of q.kT)
//      vT = (x@Wv)^T stored transposed  vT[b][feat][token]  (A operand of vT.attnT)
//    One block = 64 rows (all in one batch). 8 waves: LN then 16 tiles/wave.
//    The v-tile accumulator holds 8 consecutive tokens per lane, so the
//    transposed store is a single packed b128 store.
// ---------------------------------------------------------------------------
#define XROW 264   // padded LDS row stride in bf16 (avoids bank conflicts)
__global__ __launch_bounds__(256) void ln_kv_kernel(
    const float* __restrict__ x, const float* __restrict__ lng,
    const float* __restrict__ lnb, const bf16* __restrict__ WkT,
    const bf16* __restrict__ WvT, bf16* __restrict__ kout,
    bf16* __restrict__ vTout) {
    __shared__ bf16 xt[64 * XROW];
    int tid = threadIdx.x, lane = tid & 31, w = tid >> 5;
    size_t rowBase = (size_t)blockIdx.x * 64;
    int bglob = (int)(rowBase >> 12);        // batch index (4096 rows/batch)
    int rowInB = (int)(rowBase & 4095);      // token offset within batch

    // ---- LN phase: wave handles 8 rows, each lane 8 columns ----
    float g0[8], b0[8];
#pragma unroll
    for (int j = 0; j < 8; j++) {
        g0[j] = lng[lane * 8 + j];
        b0[j] = lnb[lane * 8 + j];
    }
    for (int i = 0; i < 8; i++) {
        int r = w * 8 + i;
        const float* px = x + (rowBase + r) * DD + lane * 8;
        float vr[8], s1 = 0.f, s2 = 0.f;
#pragma unroll
        for (int j = 0; j < 8; j++) { float t = px[j]; vr[j] = t; s1 += t; s2 += t * t; }
#pragma unroll
        for (int o = 16; o > 0; o >>= 1) {
            s1 += __shfl_xor(s1, o, 32);
            s2 += __shfl_xor(s2, o, 32);
        }
        float m = s1 * (1.f / 256.f);
        float var = s2 * (1.f / 256.f) - m * m;
        float rs = rsqrtf(var + LN_EPS);
        bf16x8 outv;
#pragma unroll
        for (int j = 0; j < 8; j++) outv[j] = (bf16)((vr[j] - m) * rs * g0[j] + b0[j]);
        *(bf16x8*)&xt[r * XROW + lane * 8] = outv;
    }
    __syncthreads();

    // ---- GEMM phase: 128 output tiles (64 for k, 64 for v), 16 per wave ----
    int halflo = (lane < 16) ? 1 : 0;
    int m16 = lane & 15;
    int aoff0 = halflo ? 0 : 16, aoff1 = halflo ? 8 : 24;
    int boff = halflo ? 0 : 16;
    for (int t = w; t < 128; t += 8) {
        int selv = t >> 6;
        int idx = t & 63;
        int mt = idx >> 4, nt = idx & 15;
        const bf16* WT = selv ? WvT : WkT;
        int ncol = nt * 16 + m16;
        v8f c = {};
#pragma unroll
        for (int kb = 0; kb < 8; kb++) {
            const bf16* ap = &xt[(mt * 16 + m16) * XROW + kb * 32];
            v16bf A = ldpair(ap + aoff0, ap + aoff1);
            const bf16* bp = WT + ncol * DD + kb * 32 + boff;
            v16bf Bv = ldpair(bp, bp + 8);
            c = wmma_bf16(A, Bv, c);
        }
        if (selv == 0) {
            // k row-major: 8 scalar b16 stores (rows = tokens, col fixed)
            size_t rbase = rowBase + mt * 16 + (halflo ? 0 : 8);
#pragma unroll
            for (int g = 0; g < 8; g++)
                kout[(rbase + g) * DD + ncol] = (bf16)c[g];
        } else {
            // vT[b][feat][token]: the 8 accumulator rows are 8 consecutive
            // tokens -> one packed 16-byte store per lane.
            bf16x8 pk;
#pragma unroll
            for (int g = 0; g < 8; g++) pk[g] = (bf16)c[g];
            size_t tok = (size_t)rowInB + mt * 16 + (halflo ? 0 : 8);
            *(bf16x8*)(vTout + ((size_t)(bglob * DD + ncol)) * NN + tok) = pk;
        }
    }
}

// ---------------------------------------------------------------------------
// 3) q = LN(slots) @ Wq  -> bf16, written into a 16-row-padded tile per batch
//    One block per (b, slot).
// ---------------------------------------------------------------------------
__global__ __launch_bounds__(256) void q_kernel(
    const float* __restrict__ slots, const float* __restrict__ lng,
    const float* __restrict__ lnb, const float* __restrict__ Wq,
    bf16* __restrict__ qout) {
    __shared__ float qin[DD];
    __shared__ float red[8];
    int bs = blockIdx.x;  // b*8 + s
    int t = threadIdx.x;
    float v = slots[bs * DD + t];
    float s1 = blockSum(v, red);
    float s2 = blockSum(v * v, red);
    float m = s1 * (1.f / 256.f);
    float var = s2 * (1.f / 256.f) - m * m;
    float rs = rsqrtf(var + LN_EPS);
    qin[t] = (v - m) * rs * lng[t] + lnb[t];
    __syncthreads();
    float acc = 0.f;
    for (int d = 0; d < DD; d++) acc += qin[d] * Wq[d * DD + t];
    int b = bs >> 3, s = bs & 7;
    qout[((size_t)b * 16 + s) * DD + t] = (bf16)acc;  // rows 8..15 stay zero
}

// ---------------------------------------------------------------------------
// 4) dots = q @ k^T * SCALE, fused softmax over the slot dim (+EPS), and
//    atomic accumulation of per-(b,slot) token sums. WMMA per 16-token tile;
//    all 8 slot logits of a token live in one lane's accumulator VGPRs.
// ---------------------------------------------------------------------------
__global__ __launch_bounds__(256) void dots_softmax_kernel(
    const bf16* __restrict__ qb, const bf16* __restrict__ kbuf,
    float* __restrict__ attn_f, float* __restrict__ attn_sum) {
    int b = blockIdx.x;
    int colblk = blockIdx.y;  // 0..7, 512 tokens each
    int tid = threadIdx.x, lane = tid & 31, w = tid >> 5;
    int halflo = (lane < 16) ? 1 : 0;
    int m16 = lane & 15;
    int aoff0 = halflo ? 0 : 16, aoff1 = halflo ? 8 : 24;
    int boff = halflo ? 0 : 16;

    // Preload A = q tile (16 slots padded x 256) into 8 K-step operands.
    v16bf A[8];
    const bf16* qrow = qb + ((size_t)b * 16 + m16) * DD;
#pragma unroll
    for (int kb = 0; kb < 8; kb++)
        A[kb] = ldpair(qrow + kb * 32 + aoff0, qrow + kb * 32 + aoff1);

    float lsum[8];
#pragma unroll
    for (int g = 0; g < 8; g++) lsum[g] = 0.f;

    int tokBase = colblk * 512 + w * 64;
    for (int tt = 0; tt < 4; tt++) {
        int tok = tokBase + tt * 16 + m16;
        const bf16* krow = kbuf + ((size_t)b * NN + tok) * DD;
        __builtin_prefetch(krow + 16 * DD, 0, 0);  // next 16-token tile
        v8f c = {};
#pragma unroll
        for (int kb = 0; kb < 8; kb++) {
            const bf16* bp = krow + kb * 32 + boff;
            v16bf Bv = ldpair(bp, bp + 8);
            c = wmma_bf16(A[kb], Bv, c);
        }
        if (halflo) {  // lanes 0..15 hold the 8 valid slot rows for this token
            float mx = -1e30f;
            float lg[8];
#pragma unroll
            for (int g = 0; g < 8; g++) { lg[g] = c[g] * SCALE; mx = fmaxf(mx, lg[g]); }
            float ssum = 0.f;
            float e[8];
#pragma unroll
            for (int g = 0; g < 8; g++) { e[g] = expf(lg[g] - mx); ssum += e[g]; }
            float inv = 1.f / ssum;
#pragma unroll
            for (int g = 0; g < 8; g++) {
                float a = e[g] * inv + EPS;
                attn_f[((size_t)(b * NSL + g)) * NN + tok] = a;
                lsum[g] += a;
            }
        }
    }
#pragma unroll
    for (int g = 0; g < 8; g++) {
        float v = lsum[g];  // lanes 16..31 contribute zero
#pragma unroll
        for (int o = 16; o > 0; o >>= 1) v += __shfl_xor(v, o, 32);
        if (lane == 0) atomicAdd(&attn_sum[b * NSL + g], v);
    }
}

// ---------------------------------------------------------------------------
// 5) attn_bf16[b][s][j] = attn_f[b][s][j] / attn_sum[b][s]   (16-row padded)
// ---------------------------------------------------------------------------
__global__ __launch_bounds__(256) void norm_kernel(
    const float* __restrict__ attn_f, const float* __restrict__ attn_sum,
    bf16* __restrict__ attn_b) {
    int idx = blockIdx.x * 256 + threadIdx.x;   // (b*8+s)*4096 + j
    int j = idx & (NN - 1);
    int bs = idx >> 12;
    int b = bs >> 3, s = bs & 7;
    float a = attn_f[idx] / attn_sum[bs];
    attn_b[((size_t)b * 16 + s) * NN + j] = (bf16)a;
}

// ---------------------------------------------------------------------------
// 6) updates^T = vT @ attn^T via WMMA (updates = attn @ v).
//    LDS-free, barrier-free pure streaming: A = vT[b][feat][tok] rows
//    (contiguous per lane), B = attn_b[b][slot][tok] rows (contiguous,
//    zero-padded slots 8..15). One block per batch; wave owns 32 features.
// ---------------------------------------------------------------------------
__global__ __launch_bounds__(256) void updates_kernel(
    const bf16* __restrict__ attn_b, const bf16* __restrict__ vT,
    float* __restrict__ updates) {
    int b = blockIdx.x, tid = threadIdx.x, lane = tid & 31, w = tid >> 5;
    int halflo = (lane < 16) ? 1 : 0;
    int m16 = lane & 15;
    int aoff0 = halflo ? 0 : 16, aoff1 = halflo ? 8 : 24;
    int boff = halflo ? 0 : 16;
    int f0 = w * 32;  // two 16-feature tiles: [f0, f0+16) and [f0+16, f0+32)
    const bf16* arow0 = vT + ((size_t)(b * DD + f0 + m16)) * NN;
    const bf16* arow1 = arow0 + (size_t)16 * NN;
    const bf16* brow = attn_b + ((size_t)(b * 16 + m16)) * NN;
    v8f c0 = {}, c1 = {};
    for (int kb = 0; kb < 128; kb++) {
        int tok = kb * 32;
        __builtin_prefetch(arow0 + tok + 128, 0, 0);
        __builtin_prefetch(arow1 + tok + 128, 0, 0);
        v16bf Bv = ldpair(brow + tok + boff, brow + tok + boff + 8);
        c0 = wmma_bf16(ldpair(arow0 + tok + aoff0, arow0 + tok + aoff1), Bv, c0);
        c1 = wmma_bf16(ldpair(arow1 + tok + aoff0, arow1 + tok + aoff1), Bv, c1);
    }
    // D tile: M = feature (VGPR idx), N = slot (lane). Slots 8..15 are padding.
    if (m16 < NSL) {
        int fo = halflo ? 0 : 8;
#pragma unroll
        for (int g = 0; g < 8; g++) {
            updates[(b * NSL + m16) * DD + f0 + fo + g] = c0[g];
            updates[(b * NSL + m16) * DD + f0 + 16 + fo + g] = c1[g];
        }
    }
}

// ---------------------------------------------------------------------------
// 7) Fused GRUCell + LN + MLP residual per (b, slot). fp32 throughout.
// ---------------------------------------------------------------------------
__global__ __launch_bounds__(256) void gru_mlp_kernel(
    float* __restrict__ slots, const float* __restrict__ updates,
    const float* __restrict__ wi, const float* __restrict__ wh,
    const float* __restrict__ bi, const float* __restrict__ bh,
    const float* __restrict__ w1, const float* __restrict__ b1,
    const float* __restrict__ w2, const float* __restrict__ b2,
    const float* __restrict__ lng, const float* __restrict__ lnb) {
    __shared__ float xrow[DD], hrow[DD], ffrow[DD], hid[HIDD], red[8];
    int bs = blockIdx.x, t = threadIdx.x;
    xrow[t] = updates[bs * DD + t];
    hrow[t] = slots[bs * DD + t];
    __syncthreads();
    float gi[3], gh[3];
#pragma unroll
    for (int cc = 0; cc < 3; cc++) {
        int row = cc * DD + t;
        const float* wir = wi + (size_t)row * DD;
        const float* whr = wh + (size_t)row * DD;
        float a1 = 0.f, a2 = 0.f;
        for (int d = 0; d < DD; d++) { a1 += xrow[d] * wir[d]; a2 += hrow[d] * whr[d]; }
        gi[cc] = a1 + bi[row];
        gh[cc] = a2 + bh[row];
    }
    float r = 1.f / (1.f + expf(-(gi[0] + gh[0])));
    float z = 1.f / (1.f + expf(-(gi[1] + gh[1])));
    float nn = tanhf(gi[2] + r * gh[2]);
    float hnew = (1.f - z) * nn + z * hrow[t];
    __syncthreads();
    float s1 = blockSum(hnew, red);
    float s2 = blockSum(hnew * hnew, red);
    float m = s1 * (1.f / 256.f);
    float var = s2 * (1.f / 256.f) - m * m;
    float rs = rsqrtf(var + LN_EPS);
    ffrow[t] = (hnew - m) * rs * lng[t] + lnb[t];
    __syncthreads();
#pragma unroll
    for (int cc = 0; cc < 2; cc++) {
        int j = cc * DD + t;
        float acc = b1[j];
        for (int d = 0; d < DD; d++) acc += ffrow[d] * w1[d * HIDD + j];
        hid[j] = fmaxf(acc, 0.f);
    }
    __syncthreads();
    float acc = b2[t];
    for (int j = 0; j < HIDD; j++) acc += hid[j] * w2[j * DD + t];
    slots[bs * DD + t] = hnew + acc;
}

// ---------------------------------------------------------------------------
// Host side
// ---------------------------------------------------------------------------
extern "C" void kernel_launch(void* const* d_in, const int* in_sizes, int n_in,
                              void* d_out, int out_size, void* d_ws, size_t ws_size,
                              hipStream_t stream) {
    const float* inputs  = (const float*)d_in[0];
    const float* noise   = (const float*)d_in[1];
    const float* mu      = (const float*)d_in[2];
    const float* lsig    = (const float*)d_in[3];
    const float* Wq      = (const float*)d_in[4];
    const float* Wk      = (const float*)d_in[5];
    const float* Wv      = (const float*)d_in[6];
    const float* gwi     = (const float*)d_in[7];
    const float* gwh     = (const float*)d_in[8];
    const float* gbi     = (const float*)d_in[9];
    const float* gbh     = (const float*)d_in[10];
    const float* w1      = (const float*)d_in[11];
    const float* b1      = (const float*)d_in[12];
    const float* w2      = (const float*)d_in[13];
    const float* b2      = (const float*)d_in[14];
    const float* ln_in_g = (const float*)d_in[15];
    const float* ln_in_b = (const float*)d_in[16];
    const float* ln_s_g  = (const float*)d_in[17];
    const float* ln_s_b  = (const float*)d_in[18];
    const float* ln_ff_g = (const float*)d_in[19];
    const float* ln_ff_b = (const float*)d_in[20];

    char* ws = (char*)d_ws;
    size_t off = 0;
    auto take = [&](size_t bytes) -> char* {
        char* p = ws + off;
        off += (bytes + 255) & ~(size_t)255;
        return p;
    };
    bf16*  kbuf     = (bf16*)take((size_t)BB * NN * DD * 2);  // 134 MB, [b][tok][f]
    bf16*  vTbuf    = (bf16*)take((size_t)BB * NN * DD * 2);  // 134 MB, [b][f][tok]
    bf16*  WkT      = (bf16*)take((size_t)DD * DD * 2);
    bf16*  WvT      = (bf16*)take((size_t)DD * DD * 2);
    float* slots    = (float*)take((size_t)BB * NSL * DD * 4);
    bf16*  qb       = (bf16*)take((size_t)BB * 16 * DD * 2);
    float* attn_f   = (float*)take((size_t)BB * NSL * NN * 4);
    bf16*  attn_b   = (bf16*)take((size_t)BB * 16 * NN * 2);
    float* attn_sum = (float*)take((size_t)BB * NSL * 4);
    float* updates  = (float*)take((size_t)BB * NSL * DD * 4);

    // Zero the padded rows (8..15) of q and attn tiles once per launch.
    hipMemsetAsync(qb, 0, (size_t)BB * 16 * DD * 2, stream);
    hipMemsetAsync(attn_b, 0, (size_t)BB * 16 * NN * 2, stream);

    prep_transpose_kernel<<<512, 256, 0, stream>>>(Wk, Wv, WkT, WvT);
    init_slots_kernel<<<(BB * NSL * DD) / 256, 256, 0, stream>>>(noise, mu, lsig, slots);
    ln_kv_kernel<<<(BB * NN) / 64, 256, 0, stream>>>(inputs, ln_in_g, ln_in_b, WkT, WvT,
                                                     kbuf, vTbuf);

    for (int it = 0; it < 4; it++) {  // ITERS=3 + final stop-gradient step
        q_kernel<<<BB * NSL, 256, 0, stream>>>(slots, ln_s_g, ln_s_b, Wq, qb);
        hipMemsetAsync(attn_sum, 0, (size_t)BB * NSL * 4, stream);
        dim3 dgrid(BB, 8);
        dots_softmax_kernel<<<dgrid, 256, 0, stream>>>(qb, kbuf, attn_f, attn_sum);
        norm_kernel<<<(BB * NSL * NN) / 256, 256, 0, stream>>>(attn_f, attn_sum, attn_b);
        updates_kernel<<<BB, 256, 0, stream>>>(attn_b, vTbuf, updates);
        gru_mlp_kernel<<<BB * NSL, 256, 0, stream>>>(slots, updates, gwi, gwh, gbi, gbh,
                                                     w1, b1, w2, b2, ln_ff_g, ln_ff_b);
    }
    hipMemcpyAsync(d_out, slots, (size_t)BB * NSL * DD * 4,
                   hipMemcpyDeviceToDevice, stream);
}
